// AttentionSTGCN_46703474377063
// MI455X (gfx1250) — compile-verified, matrix operand order checked by reading, not run on hardware
//
#include <hip/hip_runtime.h>
#include <hip/hip_bf16.h>

typedef __bf16 bf16_t;
typedef __attribute__((ext_vector_type(16))) __bf16 v16bf;
typedef __attribute__((ext_vector_type(8)))  __bf16 v8bf;
typedef __attribute__((ext_vector_type(8)))  float  v8f;

#define TDIM 256
#define HCH  64
#define LDS_STRIDE 72   // bf16 elems: 144B rows -> 16B aligned, conflict-free bank walk

// ---------------------------------------------------------------------------
// Weight prep: fp32 (L,O,I,K) -> bf16 [k][o][i] for one layer
__global__ void prep_tconv_w(const float* __restrict__ w, bf16_t* __restrict__ wt, int layer) {
    int idx = blockIdx.x * blockDim.x + threadIdx.x;
    const int total = 9 * 64 * 64;
    if (idx >= total) return;
    int i = idx % 64;
    int o = (idx / 64) % 64;
    int k = idx / (64 * 64);
    wt[(k * 64 + o) * 64 + i] = (bf16_t)w[(((layer * 64 + o) * 64) + i) * 9 + k];
}

// gcw fp32 (L, C_in, O) -> bf16 [o][i] (transposed) for one layer
__global__ void prep_gc_w(const float* __restrict__ w, bf16_t* __restrict__ wt, int layer) {
    int idx = blockIdx.x * blockDim.x + threadIdx.x;
    if (idx >= 64 * 64) return;
    int i = idx % 64;
    int o = idx / 64;
    wt[o * 64 + i] = (bf16_t)w[(layer * 64 + i) * 64 + o];
}

// ---------------------------------------------------------------------------
// Projection: kps (B,T,N,2) @ pw(2,64) + pb -> h (B,N,64,T) fp32 + bf16
// grid: (B*N, 64), block: 256 (= T)
__global__ __launch_bounds__(256)
void proj_kernel(const float* __restrict__ kps, const float* __restrict__ pw,
                 const float* __restrict__ pb, float* __restrict__ hf,
                 bf16_t* __restrict__ hb, int N, int T) {
    int bn = blockIdx.x;
    int c  = blockIdx.y;
    int t  = threadIdx.x;
    int n  = bn % N;        // scalar, once per block
    int b  = bn / N;
    float x0 = kps[((size_t)(b * T + t) * N + n) * 2 + 0];
    float x1 = kps[((size_t)(b * T + t) * N + n) * 2 + 1];
    float v = x0 * pw[c] + x1 * pw[64 + c] + pb[c];
    size_t idx = ((size_t)bn * HCH + c) * T + t;
    hf[idx] = v;
    hb[idx] = (bf16_t)v;
}

// ---------------------------------------------------------------------------
// Fused temporal-conv (TAPS=9) or channel matmul (TAPS=1) via bf16 WMMA.
// in: (BN, 64, T) bf16, wt: [tap][o][i] bf16. One wave = one 16x16 (o,t) tile.
template <int TAPS>
__global__ __launch_bounds__(128)
void conv_wmma_kernel(const bf16_t* __restrict__ in, const bf16_t* __restrict__ wt,
                      const float* __restrict__ bias, const float* __restrict__ g,
                      const float* __restrict__ be, const float* __restrict__ mn,
                      const float* __restrict__ vr, const float* __restrict__ res,
                      float* __restrict__ out_f32, bf16_t* __restrict__ out_bf,
                      int T, int relu) {
    constexpr int HALO = TAPS / 2;
    constexpr int COLS = 16 + 2 * HALO;
    // Transposed tile: xst[time][channel]; 8 K-consecutive channels per lane
    // form one aligned 128-bit ds_load.
    __shared__ __attribute__((aligned(16))) bf16_t xst[COLS][LDS_STRIDE];

    int bn = blockIdx.x;
    int t0 = blockIdx.y * 16;
    int tid = threadIdx.x;

    // Stage input tile (temporal halo, zero-padded). Consecutive tid ->
    // consecutive t -> coalesced global loads.
#pragma unroll
    for (int idx = 0; idx < 64 * COLS; idx += 128) {
        int e   = idx + tid;
        int c   = e / COLS;
        int col = e - c * COLS;
        int t = t0 - HALO + col;
        bf16_t v = (bf16_t)0.0f;
        if (t >= 0 && t < T) v = in[((size_t)bn * HCH + c) * T + t];
        if (c < 64) xst[col][c] = v;
    }
    __syncthreads();

    int wave = tid >> 5;
    int lane = tid & 31;
    int o0 = wave << 4;        // 4 waves cover 64 output channels
    int nn = lane & 15;        // time-within-tile (B cols) / out-chan (A rows)
    int kg = (lane >> 4) << 3; // K-group offset per lane half (ISA A/B layout)

    v8f acc = {};
#pragma unroll
    for (int tap = 0; tap < TAPS; ++tap) {
#pragma unroll
        for (int c0 = 0; c0 < 64; c0 += 32) {
            const bf16_t* wrow = wt + ((size_t)(tap * 64 + o0 + nn)) * 64 + c0;
            v8bf alo = *(const v8bf*)(wrow + kg);           // 16B global loads
            v8bf ahi = *(const v8bf*)(wrow + 16 + kg);
            const bf16_t* brow = &xst[nn + tap][c0];
            v8bf blo = *(const v8bf*)(brow + kg);           // 16B LDS loads
            v8bf bhi = *(const v8bf*)(brow + 16 + kg);
            v16bf a = __builtin_shufflevector(alo, ahi, 0, 1, 2, 3, 4, 5, 6, 7,
                                              8, 9, 10, 11, 12, 13, 14, 15);
            v16bf b = __builtin_shufflevector(blo, bhi, 0, 1, 2, 3, 4, 5, 6, 7,
                                              8, 9, 10, 11, 12, 13, 14, 15);
            acc = __builtin_amdgcn_wmma_f32_16x16x32_bf16(
                false, a, false, b, (short)0, acc, false, false);
        }
    }

    // Epilogue: bias + eval BatchNorm + optional ReLU + optional residual
#pragma unroll
    for (int r = 0; r < 8; ++r) {
        int o = o0 + kg + r;   // C/D layout: M = vgpr + 8*(lane>=16)
        int t = t0 + nn;
        size_t idx = ((size_t)bn * HCH + o) * T + t;
        float val = acc[r];
        if (bias) val += bias[o];
        if (g) {
            float sc = g[o] * rsqrtf(vr[o] + 1e-5f);
            val = val * sc + (be[o] - mn[o] * sc);
        }
        if (relu) val = fmaxf(val, 0.0f);
        if (res) val += res[idx];
        if (out_f32) out_f32[idx] = val;
        out_bf[idx] = (bf16_t)val;
    }
}

// ---------------------------------------------------------------------------
// Adjacency contraction: out[b,m,o,t] = relu(sum_n adj[m,n]*s[b,n,o,t] + gb[o])
// grid: (B*N, 64), block: 256 (= T)
__global__ __launch_bounds__(256)
void adj_kernel(const bf16_t* __restrict__ s, const float* __restrict__ adj,
                const float* __restrict__ gb, bf16_t* __restrict__ out,
                int N, int T) {
    int bm = blockIdx.x;
    int o  = blockIdx.y;
    int t  = threadIdx.x;
    int m  = bm % N;        // scalar, once per block
    int b  = bm / N;
    float accv = gb[o];
    const float* arow = adj + m * N;
    const bf16_t* srow = s + (((size_t)b * N) * HCH + o) * T + t;
    for (int n = 0; n < N; ++n) {
        float av = arow[n];
        if (av != 0.0f) accv += av * (float)srow[(size_t)n * HCH * T];
    }
    out[((size_t)bm * HCH + o) * T + t] = (bf16_t)fmaxf(accv, 0.0f);
}

// ---------------------------------------------------------------------------
__global__ void pool_kernel(const float* __restrict__ a, float* __restrict__ feat,
                            int Bn, int N, int T, int off) {
    int idx = blockIdx.x * blockDim.x + threadIdx.x;
    int total = Bn * N * HCH;
    if (idx >= total) return;
    int c = idx % HCH;
    int n = (idx / HCH) % N;
    int b = idx / (HCH * N);
    const float* p = a + (((size_t)b * N + n) * HCH + c) * T;
    float s = 0.0f;
    for (int t = 0; t < T; ++t) s += p[t];
    feat[(size_t)b * 5504 + off + n * HCH + c] = s / (float)T;
}

__global__ void head_kernel(const float* __restrict__ hp, const float* __restrict__ w,
                            const float* __restrict__ bias, float* __restrict__ feat,
                            int off) {
    int idx = blockIdx.x * blockDim.x + threadIdx.x;
    if (idx >= 8 * HCH) return;
    int c = idx % HCH;
    int b = idx / HCH;
    float s = 0.0f;
    for (int t = 0; t < TDIM; ++t) {
        float a = bias[c];
        for (int p = 0; p < 6; ++p) a += hp[((size_t)(b * TDIM + t)) * 6 + p] * w[p * HCH + c];
        s += fmaxf(a, 0.0f);
    }
    feat[(size_t)b * 5504 + off + c] = s / (float)TDIM;
}

__global__ void dense_kernel(const float* __restrict__ x, const float* __restrict__ w,
                             const float* __restrict__ bias, float* __restrict__ y,
                             int Bn, int In, int Out, int relu) {
    int idx = blockIdx.x * blockDim.x + threadIdx.x;
    if (idx >= Bn * Out) return;
    int o = idx % Out;
    int b = idx / Out;
    float s = bias[o];
    const float* xr = x + (size_t)b * In;
    for (int i = 0; i < In; ++i) s += xr[i] * w[(size_t)i * Out + o];
    y[idx] = relu ? fmaxf(s, 0.0f) : s;
}

// ---------------------------------------------------------------------------
extern "C" void kernel_launch(void* const* d_in, const int* in_sizes, int n_in,
                              void* d_out, int out_size, void* d_ws, size_t ws_size,
                              hipStream_t stream) {
    (void)in_sizes; (void)n_in; (void)out_size; (void)ws_size;
    const int B = 8, T = TDIM, NB = 17, NF = 68, L = 3;

    const float* body_kps = (const float*)d_in[0];
    const float* face_kps = (const float*)d_in[1];
    const float* head_pose = (const float*)d_in[2];
    const float* body_adj = (const float*)d_in[3];
    const float* face_adj = (const float*)d_in[4];
    const float* body_pw = (const float*)d_in[5];
    const float* body_pb = (const float*)d_in[6];
    const float* face_pw = (const float*)d_in[7];
    const float* face_pb = (const float*)d_in[8];
    const float* head_w = (const float*)d_in[9];
    const float* head_b = (const float*)d_in[10];

    // Workspace layout
    char* base = (char*)d_ws;
    size_t off = 0;
    auto take = [&](size_t bytes) {
        char* p = base + off;
        off = (off + bytes + 255) & ~(size_t)255;
        return (void*)p;
    };
    const size_t ACT = (size_t)B * NF * HCH * T;  // max (face) activation elems
    float*  A_f32 = (float*)take(ACT * 4);
    bf16_t* A_bf  = (bf16_t*)take(ACT * 2);
    bf16_t* B_bf  = (bf16_t*)take(ACT * 2);
    bf16_t* C_bf  = (bf16_t*)take(ACT * 2);
    bf16_t* wt1   = (bf16_t*)take((size_t)9 * 64 * 64 * 2);
    bf16_t* wt2   = (bf16_t*)take((size_t)9 * 64 * 64 * 2);
    bf16_t* wg    = (bf16_t*)take((size_t)64 * 64 * 2);
    float*  feat  = (float*)take((size_t)B * 5504 * 4);
    float*  m1    = (float*)take((size_t)B * 256 * 4);
    float*  m2    = (float*)take((size_t)B * 128 * 4);
    float*  m3    = (float*)take((size_t)B * 64 * 4);

    for (int s = 0; s < 2; ++s) {
        int N = s == 0 ? NB : NF;
        const float* kps = s == 0 ? body_kps : face_kps;
        const float* pw = s == 0 ? body_pw : face_pw;
        const float* pb = s == 0 ? body_pb : face_pb;
        const float* adj = s == 0 ? body_adj : face_adj;
        const int pbase = s == 0 ? 11 : 25;
        const float* t1w  = (const float*)d_in[pbase + 0];
        const float* t1b  = (const float*)d_in[pbase + 1];
        const float* bn1g = (const float*)d_in[pbase + 2];
        const float* bn1b = (const float*)d_in[pbase + 3];
        const float* bn1m = (const float*)d_in[pbase + 4];
        const float* bn1v = (const float*)d_in[pbase + 5];
        const float* gcw  = (const float*)d_in[pbase + 6];
        const float* gcb  = (const float*)d_in[pbase + 7];
        const float* t2w  = (const float*)d_in[pbase + 8];
        const float* t2b  = (const float*)d_in[pbase + 9];
        const float* bn2g = (const float*)d_in[pbase + 10];
        const float* bn2b = (const float*)d_in[pbase + 11];
        const float* bn2m = (const float*)d_in[pbase + 12];
        const float* bn2v = (const float*)d_in[pbase + 13];

        int BN = B * N;
        dim3 egrid(BN, HCH);   // elementwise kernels: block = T threads
        dim3 cgrid(BN, T / 16);

        proj_kernel<<<egrid, T, 0, stream>>>(kps, pw, pb, A_f32, A_bf, N, T);

        for (int l = 0; l < L; ++l) {
            prep_tconv_w<<<(9 * 64 * 64 + 255) / 256, 256, 0, stream>>>(t1w, wt1, l);
            prep_tconv_w<<<(9 * 64 * 64 + 255) / 256, 256, 0, stream>>>(t2w, wt2, l);
            prep_gc_w<<<(64 * 64 + 255) / 256, 256, 0, stream>>>(gcw, wg, l);

            // conv1 + BN + ReLU  (bf16 out only)
            conv_wmma_kernel<9><<<cgrid, 128, 0, stream>>>(
                A_bf, wt1, t1b + l * 64, bn1g + l * 64, bn1b + l * 64,
                bn1m + l * 64, bn1v + l * 64, nullptr, nullptr, B_bf, T, 1);
            // channel matmul (graph-conv support), taps=1, plain
            conv_wmma_kernel<1><<<cgrid, 128, 0, stream>>>(
                B_bf, wg, nullptr, nullptr, nullptr, nullptr, nullptr, nullptr,
                nullptr, C_bf, T, 0);
            // adjacency + gb + ReLU
            adj_kernel<<<egrid, T, 0, stream>>>(C_bf, adj, gcb + l * 64, B_bf, N, T);
            // conv2 + BN + residual (fp32 master + bf16 shadow, in-place on A)
            conv_wmma_kernel<9><<<cgrid, 128, 0, stream>>>(
                B_bf, wt2, t2b + l * 64, bn2g + l * 64, bn2b + l * 64,
                bn2m + l * 64, bn2v + l * 64, A_f32, A_f32, A_bf, T, 0);
        }
        int featOff = s == 0 ? 0 : NB * HCH;
        pool_kernel<<<(BN * HCH + 255) / 256, 256, 0, stream>>>(A_f32, feat, B, N, T, featOff);
    }

    head_kernel<<<2, 256, 0, stream>>>(head_pose, head_w, head_b, feat, (NB + NF) * HCH);

    const float* fus1_w = (const float*)d_in[39];
    const float* fus1_b = (const float*)d_in[40];
    const float* fus2_w = (const float*)d_in[41];
    const float* fus2_b = (const float*)d_in[42];
    const float* fus3_w = (const float*)d_in[43];
    const float* fus3_b = (const float*)d_in[44];
    const float* fus4_w = (const float*)d_in[45];
    const float* fus4_b = (const float*)d_in[46];

    dense_kernel<<<(8 * 256 + 255) / 256, 256, 0, stream>>>(feat, fus1_w, fus1_b, m1, 8, 5504, 256, 1);
    dense_kernel<<<(8 * 128 + 255) / 256, 256, 0, stream>>>(m1, fus2_w, fus2_b, m2, 8, 256, 128, 1);
    dense_kernel<<<(8 * 64 + 255) / 256, 256, 0, stream>>>(m2, fus3_w, fus3_b, m3, 8, 128, 64, 1);
    dense_kernel<<<1, 64, 0, stream>>>(m3, fus4_w, fus4_b, (float*)d_out, 8, 64, 5, 0);
}
